// SpatioTemporalBlock_89781996356052
// MI455X (gfx1250) — matrix operand order repaired and checked by reading, not compiled, not run on hardware
//
#include <hip/hip_runtime.h>
#include <hip/hip_bf16.h>

// ---------------------------------------------------------------------------
// CDNA5 (gfx1250) fused SpatioTemporalBlock.
// Matrix-core bound (~86 GFLOP vs ~200MB HBM) -> all GEMMs via
// v_wmma_f32_16x16x32_f16, f32 accumulate, LDS-resident tiles.
// Round 2: stage K3's strided H_p tile with the Tensor Data Mover
// (tensor_load_to_lds + s_wait_tensorcnt) instead of scalar ds stores.
// ---------------------------------------------------------------------------

typedef __attribute__((ext_vector_type(16))) _Float16 v16h;
typedef __attribute__((ext_vector_type(8)))  _Float16 v8h;
typedef __attribute__((ext_vector_type(8)))  float    v8f;

typedef __attribute__((ext_vector_type(4))) unsigned int tdm_g0_t;
typedef __attribute__((ext_vector_type(8))) int          tdm_g1_t;
typedef __attribute__((ext_vector_type(4))) int          tdm_g2_t;

#define DEV __device__ __forceinline__

// ----- WMMA fragment loaders (wave32; contiguous K per lane half, same order in A and B) ---------
// A 16x32 f16: lanes 0-15 -> M=lane, halfs 0..15 = K 0..15
//              lanes 16-31 -> M=lane-16, halfs = K 16..31
DEV v16h load_a(const _Float16* __restrict__ A, int lda) {
  const int lane = threadIdx.x & 31;
  const int g = lane >> 4;
  const int r = lane & 15;
  const _Float16* p0 = A + r * lda + g * 16;
  v8h lo = *(const v8h*)p0;
  v8h hi = *(const v8h*)(p0 + 8);
  v16h out;
#pragma unroll
  for (int i = 0; i < 8; ++i) { out[i] = lo[i]; out[i + 8] = hi[i]; }
  return out;
}

// B 32x16 f16, supplied as B^T (N x K row-major): lane n = lane&15,
// halfs e -> K = (lane>>4)*16 + e  => one contiguous 32B load per lane.
DEV v16h load_bt(const _Float16* __restrict__ Bt, int ldb) {
  const int lane = threadIdx.x & 31;
  const int g = lane >> 4;
  const int n = lane & 15;
  return *(const v16h*)(Bt + n * ldb + g * 16);
}

// Generic wave-tiled GEMM: OUT(MxN) = A(MxK) @ B(KxN), B given transposed.
// Each wave owns 16x16 output tiles round-robin; epilogue handles
// bias/activation/residual/transpose-store. EXEC stays full (no divergence).
template <typename Epi>
DEV void gemm_bt(const _Float16* __restrict__ A, int lda,
                 const _Float16* __restrict__ Bt, int ldb,
                 int M, int N, int K, int wave, int nwaves, Epi&& epi)
{
  const int tn = N >> 4;
  const int tiles = (M >> 4) * tn;
  for (int t = wave; t < tiles; t += nwaves) {
    const int m0 = (t / tn) << 4;
    const int n0 = (t % tn) << 4;
    v8f acc = {0.f, 0.f, 0.f, 0.f, 0.f, 0.f, 0.f, 0.f};
    for (int k = 0; k < K; k += 32) {
      v16h a = load_a(A + m0 * lda + k, lda);
      v16h b = load_bt(Bt + n0 * ldb + k, ldb);
      acc = __builtin_amdgcn_wmma_f32_16x16x32_f16(
          false, a, false, b, (short)0, acc, false, false);
    }
    epi(m0, n0, acc);
  }
}

// ---------------------------------------------------------------------------
// K0: f32 -> f16 weight conversion (layouts kept: W[n][k] == B^T row-major)
// ---------------------------------------------------------------------------
__global__ void k0_convert(const float* __restrict__ src,
                           _Float16* __restrict__ dst, int n)
{
  for (int i = blockIdx.x * blockDim.x + threadIdx.x; i < n;
       i += gridDim.x * blockDim.x)
    dst[i] = (_Float16)src[i];
}

// ---------------------------------------------------------------------------
// K1: per-(b,c) temporal transformer. 1024 blocks x 256 threads (8 waves).
// LDS pool (128KB):
//   [0,32K)      sO   : attention output f16 / FFN t1 chunk f16
//   [32K,64K)    sH   : ln1 activations f16        (stage A)
//   [64K,96K)    sS   : scores / softmax f16       (stage A)
//   [32K,96K)    sX   : residual stream f32        (stage B, overlays sH+sS)
//   [96K,128K)   sQ/sK/sVT (stage A)  |  sH2 ln2 f16 (stage B)
// ---------------------------------------------------------------------------
__global__ void k1_temporal(const float* __restrict__ Hin,
    const _Float16* __restrict__ Wqkv, const float* __restrict__ bqkv,
    const _Float16* __restrict__ Wout, const float* __restrict__ bout,
    const float* __restrict__ ln1g, const float* __restrict__ ln1b,
    const float* __restrict__ ln2g, const float* __restrict__ ln2b,
    const _Float16* __restrict__ Wff1, const float* __restrict__ bff1,
    const _Float16* __restrict__ Wff2, const float* __restrict__ bff2,
    const float* __restrict__ tng, const float* __restrict__ tnb,
    float* __restrict__ Htemp, _Float16* __restrict__ H16,
    float* __restrict__ Hs)
{
  extern __shared__ __align__(32) char pool[];
  _Float16* sO  = (_Float16*)(pool);
  _Float16* sH  = (_Float16*)(pool + 32768);
  _Float16* sS  = (_Float16*)(pool + 65536);
  float*    sX  = (float*)   (pool + 32768);
  _Float16* sQ  = (_Float16*)(pool + 98304);
  _Float16* sK  = (_Float16*)(pool + 106496);
  _Float16* sVT = (_Float16*)(pool + 114688);
  _Float16* sH2 = (_Float16*)(pool + 98304);
  _Float16* sT1 = sO;

  const int bc   = blockIdx.x;
  const int tid  = threadIdx.x;
  const int wave = tid >> 5;
  const int NW   = 8;
  const float* x0 = Hin + (size_t)bc * 16384;

  __builtin_prefetch(Wqkv, 0, 1);
  __builtin_prefetch(Wff1, 0, 1);

  // ---- LN1 -> sH (f16) ----
  for (int r = tid; r < 128; r += 256) {
    const float* xr = x0 + r * 128;
    float m = 0.f;
    for (int d = 0; d < 128; ++d) m += xr[d];
    m *= (1.f / 128.f);
    float v = 0.f;
    for (int d = 0; d < 128; ++d) { float t = xr[d] - m; v += t * t; }
    v *= (1.f / 128.f);
    float rs = rsqrtf(v + 1e-5f);
    for (int d = 0; d < 128; ++d)
      sH[r * 128 + d] = (_Float16)((xr[d] - m) * rs * ln1g[d] + ln1b[d]);
  }
  __syncthreads();

  // ---- multi-head attention (HT=4, dh=32) ----
  for (int ht = 0; ht < 4; ++ht) {
    gemm_bt(sH, 128, Wqkv + (ht * 32) * 128, 128, 128, 32, 128, wave, NW,
      [&](int m0, int n0, v8f acc) {
        const int lane = threadIdx.x & 31, g = lane >> 4, nn = lane & 15;
#pragma unroll
        for (int vv = 0; vv < 8; ++vv) {
          int m = m0 + vv + 8 * g, n = n0 + nn;
          sQ[m * 32 + n] = (_Float16)(acc[vv] + bqkv[ht * 32 + n]);
        }
      });
    gemm_bt(sH, 128, Wqkv + (128 + ht * 32) * 128, 128, 128, 32, 128, wave, NW,
      [&](int m0, int n0, v8f acc) {
        const int lane = threadIdx.x & 31, g = lane >> 4, nn = lane & 15;
#pragma unroll
        for (int vv = 0; vv < 8; ++vv) {
          int m = m0 + vv + 8 * g, n = n0 + nn;
          sK[m * 32 + n] = (_Float16)(acc[vv] + bqkv[128 + ht * 32 + n]);
        }
      });
    gemm_bt(sH, 128, Wqkv + (256 + ht * 32) * 128, 128, 128, 32, 128, wave, NW,
      [&](int m0, int n0, v8f acc) {
        const int lane = threadIdx.x & 31, g = lane >> 4, nn = lane & 15;
#pragma unroll
        for (int vv = 0; vv < 8; ++vv) {
          int m = m0 + vv + 8 * g, n = n0 + nn;
          sVT[n * 128 + m] = (_Float16)(acc[vv] + bqkv[256 + ht * 32 + n]);  // v^T
        }
      });
    __syncthreads();

    // scores S = q @ k^T / sqrt(dh)   (B^T = k row-major, ld 32)
    gemm_bt(sQ, 32, sK, 32, 128, 128, 32, wave, NW,
      [&](int m0, int n0, v8f acc) {
        const int lane = threadIdx.x & 31, g = lane >> 4, nn = lane & 15;
#pragma unroll
        for (int vv = 0; vv < 8; ++vv)
          sS[(m0 + vv + 8 * g) * 128 + n0 + nn] =
              (_Float16)(acc[vv] * 0.1767766953f);
      });
    __syncthreads();

    // row softmax in place (each thread owns a row -> no races)
    for (int r = tid; r < 128; r += 256) {
      _Float16* row = sS + r * 128;
      float mx = -1e30f;
      for (int j = 0; j < 128; ++j) { float t = (float)row[j]; mx = t > mx ? t : mx; }
      float sum = 0.f;
      for (int j = 0; j < 128; ++j) {
        float e = __expf((float)row[j] - mx);
        sum += e; row[j] = (_Float16)e;
      }
      float inv = 1.f / sum;
      for (int j = 0; j < 128; ++j) row[j] = (_Float16)((float)row[j] * inv);
    }
    __syncthreads();

    // O_h = att @ v  (B^T = sVT, ld 128) -> sO columns [ht*32, ht*32+32)
    gemm_bt(sS, 128, sVT, 128, 128, 32, 128, wave, NW,
      [&](int m0, int n0, v8f acc) {
        const int lane = threadIdx.x & 31, g = lane >> 4, nn = lane & 15;
#pragma unroll
        for (int vv = 0; vv < 8; ++vv)
          sO[(m0 + vv + 8 * g) * 128 + ht * 32 + n0 + nn] = (_Float16)acc[vv];
      });
    __syncthreads();
  }

  // ---- x = x0 + O @ Wout^T + b ----
  gemm_bt(sO, 128, Wout, 128, 128, 128, 128, wave, NW,
    [&](int m0, int n0, v8f acc) {
      const int lane = threadIdx.x & 31, g = lane >> 4, nn = lane & 15;
#pragma unroll
      for (int vv = 0; vv < 8; ++vv) {
        int m = m0 + vv + 8 * g, n = n0 + nn;
        sX[m * 128 + n] = x0[m * 128 + n] + acc[vv] + bout[n];
      }
    });
  __syncthreads();

  // ---- LN2 -> sH2 ----
  for (int r = tid; r < 128; r += 256) {
    const float* xr = sX + r * 128;
    float m = 0.f;
    for (int d = 0; d < 128; ++d) m += xr[d];
    m *= (1.f / 128.f);
    float v = 0.f;
    for (int d = 0; d < 128; ++d) { float t = xr[d] - m; v += t * t; }
    v *= (1.f / 128.f);
    float rs = rsqrtf(v + 1e-5f);
    for (int d = 0; d < 128; ++d)
      sH2[r * 128 + d] = (_Float16)((xr[d] - m) * rs * ln2g[d] + ln2b[d]);
  }
  __syncthreads();

  // ---- FFN in four 128-wide chunks: x += relu(h2@ff1^T)@ff2^T + b2 ----
  for (int cc = 0; cc < 4; ++cc) {
    gemm_bt(sH2, 128, Wff1 + cc * 128 * 128, 128, 128, 128, 128, wave, NW,
      [&](int m0, int n0, v8f acc) {
        const int lane = threadIdx.x & 31, g = lane >> 4, nn = lane & 15;
#pragma unroll
        for (int vv = 0; vv < 8; ++vv) {
          int m = m0 + vv + 8 * g, n = n0 + nn;
          float t = acc[vv] + bff1[cc * 128 + n];
          sT1[m * 128 + n] = (_Float16)(t > 0.f ? t : 0.f);
        }
      });
    __syncthreads();
    gemm_bt(sT1, 128, Wff2 + cc * 128, 512, 128, 128, 128, wave, NW,
      [&](int m0, int n0, v8f acc) {
        const int lane = threadIdx.x & 31, g = lane >> 4, nn = lane & 15;
#pragma unroll
        for (int vv = 0; vv < 8; ++vv) {
          int m = m0 + vv + 8 * g, n = n0 + nn;
          sX[m * 128 + n] += acc[vv] + (cc == 0 ? bff2[n] : 0.f);
        }
      });
    __syncthreads();
  }

  // ---- H_temp = tnorm(x0 + x); also keep in sX for the Hs reduction ----
  for (int r = tid; r < 128; r += 256) {
    const float* xr = x0 + r * 128;
    float* sxr = sX + r * 128;
    float m = 0.f;
    for (int d = 0; d < 128; ++d) m += xr[d] + sxr[d];
    m *= (1.f / 128.f);
    float v = 0.f;
    for (int d = 0; d < 128; ++d) { float t = xr[d] + sxr[d] - m; v += t * t; }
    v *= (1.f / 128.f);
    float rs = rsqrtf(v + 1e-5f);
    for (int d = 0; d < 128; ++d) {
      float t = (xr[d] + sxr[d] - m) * rs * tng[d] + tnb[d];
      sxr[d] = t;
      Htemp[(size_t)bc * 16384 + r * 128 + d] = t;
      H16[(size_t)bc * 16384 + r * 128 + d] = (_Float16)t;
    }
  }
  __syncthreads();
  for (int d = tid; d < 128; d += 256) {
    float s = 0.f;
    for (int p = 0; p < 128; ++p) s += sX[p * 128 + d];
    Hs[(size_t)bc * 128 + d] = s * (1.f / 128.f);
  }
}

// ---------------------------------------------------------------------------
// K2: per-batch graph scoring + top-8 mask. 16 blocks x 256 threads.
// ---------------------------------------------------------------------------
__global__ void k2_graph(const float* __restrict__ Hs,
                         const float* __restrict__ sadj,
                         const float* __restrict__ qw, const float* __restrict__ qb,
                         const float* __restrict__ kw, const float* __restrict__ kb,
                         float* __restrict__ maskf)
{
  extern __shared__ __align__(32) char pool[];
  float* sHs = (float*)pool;              // 64*128
  float* sq  = (float*)(pool + 32768);    // 64*64
  float* sk  = (float*)(pool + 49152);    // 64*64
  float* sA  = (float*)(pool + 65536);    // 64*64
  const int b = blockIdx.x, tid = threadIdx.x;

  for (int i = tid; i < 64 * 128; i += 256) sHs[i] = Hs[(size_t)b * 8192 + i];
  __syncthreads();
  for (int i = tid; i < 4096; i += 256) {
    int c = i >> 6, j = i & 63;
    float s = 0.f;
    for (int d = 0; d < 128; ++d) s += sHs[c * 128 + d] * qw[j * 128 + d];
    sq[i] = s + qb[j];
  }
  for (int i = tid; i < 4096; i += 256) {
    int c = i >> 6, j = i & 63;
    float s = 0.f;
    for (int d = 0; d < 128; ++d) s += sHs[c * 128 + d] * kw[j * 128 + d];
    sk[i] = s + kb[j];
  }
  __syncthreads();
  for (int i = tid; i < 4096; i += 256) {
    int c = i >> 6, e = i & 63;
    float s = 0.f;
    for (int d = 0; d < 64; ++d) s += sq[c * 64 + d] * sk[e * 64 + d];
    sA[i] = tanhf(s * 0.125f) + sadj[c * 64 + e];
  }
  __syncthreads();
  if (tid < 64) {
    const int c = tid;
    const float* row = sA + c * 64;
    bool chosen[64];
    for (int e = 0; e < 64; ++e) chosen[e] = false;
    for (int it = 0; it < 8; ++it) {          // TOPK = 8
      float best = -1e38f; int bi = 0;
      for (int e = 0; e < 64; ++e)
        if (!chosen[e] && row[e] > best) { best = row[e]; bi = e; }
      chosen[bi] = true;
    }
    for (int e = 0; e < 64; ++e) {
      float m = (chosen[e] && e != c && row[e] != 0.f) ? 1.f : 0.f;
      maskf[((size_t)b * 64 + c) * 64 + e] = m;
    }
  }
}

// ---------------------------------------------------------------------------
// K3: per-(b,p) GAT + head-mean + residual + snorm LN. 2048 blocks.
// LDS: Hp16 16K | XhT 16K | aT 8K | outT f32 32K | asrc/adst 512B
// H_p staging uses the Tensor Data Mover: one 2D descriptor
// (2B elems, tile 128x64, tensor row stride Np*D=16384) issued by wave 0.
// ---------------------------------------------------------------------------
__global__ void k3_gat(const float* __restrict__ Htemp,
                       const _Float16* __restrict__ H16,
                       const _Float16* __restrict__ Wgat,
                       const float* __restrict__ att_src,
                       const float* __restrict__ att_dst,
                       const float* __restrict__ gbias,
                       const float* __restrict__ maskf,
                       const float* __restrict__ sng,
                       const float* __restrict__ snb,
                       float* __restrict__ out)
{
  extern __shared__ __align__(32) char pool[];
  _Float16* sHp  = (_Float16*)(pool);           // 64 x 128
  _Float16* sXhT = (_Float16*)(pool + 16384);   // 128 x 64 (Xh transposed)
  _Float16* sAT  = (_Float16*)(pool + 32768);   // 64 x 64 (a transposed)
  float*    sOutT= (float*)   (pool + 40960);   // 128 x 64 f32
  float*    sSrc = (float*)   (pool + 73728);   // 64
  float*    sDst = (float*)   (pool + 73984);   // 64

  const int b = blockIdx.x >> 7;
  const int p = blockIdx.x & 127;
  const int tid = threadIdx.x;
  const int wave = tid >> 5;
  const int NW = 8;

  // ---- TDM: DMA the strided 64x128-half tile H16[b, :, p, :] into sHp ----
  if (wave == 0) {
    unsigned long long ga =
        (unsigned long long)(uintptr_t)(H16 + ((size_t)b * 64 * 128 + p) * 128);
    tdm_g0_t g0;
    g0[0] = 1u;                                   // D# group0: count=1 (valid)
    g0[1] = 0u;                                   // lds_addr = pool base (sHp)
    g0[2] = (unsigned)(ga & 0xffffffffu);         // global_addr[31:0]
    g0[3] = (unsigned)(((ga >> 32) & 0x1ffffffu) | (2u << 30)); // [56:32]|type=2
    tdm_g1_t g1;
    g1[0] = (int)(1u << 16);                      // data_size=1 -> 2 bytes
    g1[1] = (int)(128u << 16);                    // tensor_dim0 = 128 (low 16)
    g1[2] = (int)(64u << 16);                     // tensor_dim1 = 64
    g1[3] = (int)(128u << 16);                    // tile_dim0 = 128
    g1[4] = 64;                                   // tile_dim1 = 64, tile_dim2=0
    g1[5] = 16384;                                // tensor_dim0_stride = Np*D
    g1[6] = 0;
    g1[7] = 0;                                    // dim1 stride unused (2D)
    tdm_g2_t g2 = {0, 0, 0, 0};                   // 2D tensor: groups 2/3 zero
    tdm_g2_t g3 = {0, 0, 0, 0};
#if __clang_major__ >= 23
    tdm_g1_t gx = {0, 0, 0, 0, 0, 0, 0, 0};
    __builtin_amdgcn_tensor_load_to_lds(g0, g1, g2, g3, gx, 0);
#else
    __builtin_amdgcn_tensor_load_to_lds(g0, g1, g2, g3, 0);
#endif
    __builtin_amdgcn_s_wait_tensorcnt(0);         // TENSORcnt==0: tile in LDS
  }
  __syncthreads();

  for (int h = 0; h < 4; ++h) {
    // Xh[c,f] = H_p @ gat_w[h]^T  (B^T = gat_w[h], F x D row-major)
    gemm_bt(sHp, 128, Wgat + h * 16384, 128, 64, 128, 128, wave, NW,
      [&](int m0, int n0, v8f acc) {
        const int lane = threadIdx.x & 31, g = lane >> 4, nn = lane & 15;
#pragma unroll
        for (int vv = 0; vv < 8; ++vv)
          sXhT[(n0 + nn) * 64 + (m0 + vv + 8 * g)] = (_Float16)acc[vv];
      });
    __syncthreads();

    // asrc/adst dot products
    for (int c = tid; c < 64; c += 256) {
      float s1 = 0.f, s2 = 0.f;
      for (int f = 0; f < 128; ++f) {
        float x = (float)sXhT[f * 64 + c];
        s1 += x * att_src[h * 128 + f];
        s2 += x * att_dst[h * 128 + f];
      }
      sSrc[c] = s1; sDst[c] = s2;
    }
    __syncthreads();

    // masked leaky-relu softmax over sources s for each target t -> aT[t][s]
    for (int t = tid; t < 64; t += 256) {
      float adt = sDst[t];
      float mx = -1e30f;
      for (int s = 0; s < 64; ++s) {
        float mk = maskf[((size_t)b * 64 + s) * 64 + t];
        if (mk != 0.f) {
          float e = sSrc[s] + adt; e = e > 0.f ? e : 0.2f * e;
          mx = e > mx ? e : mx;
        }
      }
      float sum = 0.f;
      for (int s = 0; s < 64; ++s) {
        float mk = maskf[((size_t)b * 64 + s) * 64 + t];
        float w = 0.f;
        if (mk != 0.f) {
          float e = sSrc[s] + adt; e = e > 0.f ? e : 0.2f * e;
          w = __expf(e - mx);
        }
        sAT[t * 64 + s] = (_Float16)w; sum += w;
      }
      float inv = (sum > 0.f) ? 1.f / sum : 1.f;
      for (int s = 0; s < 64; ++s)
        sAT[t * 64 + s] = (_Float16)((float)sAT[t * 64 + s] * inv);
    }
    __syncthreads();

    // outT[f,t] (+)= Xh^T @ a   (A = XhT 128x64, B^T = aT 64x64)
    gemm_bt(sXhT, 64, sAT, 64, 128, 64, 64, wave, NW,
      [&](int m0, int n0, v8f acc) {
        const int lane = threadIdx.x & 31, g = lane >> 4, nn = lane & 15;
#pragma unroll
        for (int vv = 0; vv < 8; ++vv) {
          float* d = &sOutT[(m0 + vv + 8 * g) * 64 + n0 + nn];
          if (h == 0) *d = acc[vv]; else *d += acc[vv];
        }
      });
    __syncthreads();
  }

  // final: mean over heads (/4) + bias + residual + snorm LN
  for (int t = tid; t < 64; t += 256) {
    const float* hrow = Htemp + (((size_t)b * 64 + t) * 128 + p) * 128;
    float m = 0.f;
    for (int f = 0; f < 128; ++f)
      m += sOutT[f * 64 + t] * 0.25f + gbias[f] + hrow[f];
    m *= (1.f / 128.f);
    float v = 0.f;
    for (int f = 0; f < 128; ++f) {
      float val = sOutT[f * 64 + t] * 0.25f + gbias[f] + hrow[f] - m;
      v += val * val;
    }
    v *= (1.f / 128.f);
    float rs = rsqrtf(v + 1e-5f);
    float* orow = out + (((size_t)b * 64 + t) * 128 + p) * 128;
    for (int f = 0; f < 128; ++f) {
      float val = sOutT[f * 64 + t] * 0.25f + gbias[f] + hrow[f];
      orow[f] = (val - m) * rs * sng[f] + snb[f];
    }
  }
}

// ---------------------------------------------------------------------------
// Workspace layout (bytes):
//   0        Wqkv16 (98304)   98304  Wout16 (32768)   131072 Wff1 (131072)
//   262144   Wff2   (131072)  393216 Wgat   (131072)
//   524288   Htemp f32 (67108864)
//   67633152 H16  f16  (33554432)
//   101187584 Hs  f32  (524288)
//   101711872 maskf f32 (262144)      total ~102 MB
// ---------------------------------------------------------------------------
extern "C" void kernel_launch(void* const* d_in, const int* in_sizes, int n_in,
                              void* d_out, int out_size, void* d_ws, size_t ws_size,
                              hipStream_t stream)
{
  (void)in_sizes; (void)n_in; (void)out_size; (void)ws_size;
  const float* Hin  = (const float*)d_in[0];
  const float* sadj = (const float*)d_in[1];
  const float* aiw  = (const float*)d_in[2];
  const float* aib  = (const float*)d_in[3];
  const float* aow  = (const float*)d_in[4];
  const float* aob  = (const float*)d_in[5];
  const float* ln1g = (const float*)d_in[6];
  const float* ln1b = (const float*)d_in[7];
  const float* ln2g = (const float*)d_in[8];
  const float* ln2b = (const float*)d_in[9];
  const float* f1w  = (const float*)d_in[10];
  const float* f1b  = (const float*)d_in[11];
  const float* f2w  = (const float*)d_in[12];
  const float* f2b  = (const float*)d_in[13];
  const float* tng  = (const float*)d_in[14];
  const float* tnb  = (const float*)d_in[15];
  const float* qw   = (const float*)d_in[16];
  const float* qb   = (const float*)d_in[17];
  const float* kw   = (const float*)d_in[18];
  const float* kb   = (const float*)d_in[19];
  const float* gw   = (const float*)d_in[20];
  const float* gas  = (const float*)d_in[21];
  const float* gad  = (const float*)d_in[22];
  const float* gb   = (const float*)d_in[23];
  const float* sng  = (const float*)d_in[24];
  const float* snb  = (const float*)d_in[25];

  char* ws = (char*)d_ws;
  _Float16* Wqkv16 = (_Float16*)(ws + 0);
  _Float16* Wout16 = (_Float16*)(ws + 98304);
  _Float16* Wff116 = (_Float16*)(ws + 131072);
  _Float16* Wff216 = (_Float16*)(ws + 262144);
  _Float16* Wgat16 = (_Float16*)(ws + 393216);
  float*    Htemp  = (float*)   (ws + 524288);
  _Float16* H16    = (_Float16*)(ws + 67633152);
  float*    Hs     = (float*)   (ws + 101187584);
  float*    maskf  = (float*)   (ws + 101711872);

  k0_convert<<<96, 256, 0, stream>>>(aiw, Wqkv16, 49152);
  k0_convert<<<64, 256, 0, stream>>>(aow, Wout16, 16384);
  k0_convert<<<128, 256, 0, stream>>>(f1w, Wff116, 65536);
  k0_convert<<<128, 256, 0, stream>>>(f2w, Wff216, 65536);
  k0_convert<<<128, 256, 0, stream>>>(gw, Wgat16, 65536);

  k1_temporal<<<1024, 256, 131072, stream>>>(
      Hin, Wqkv16, aib, Wout16, aob, ln1g, ln1b, ln2g, ln2b,
      Wff116, f1b, Wff216, f2b, tng, tnb, Htemp, H16, Hs);

  k2_graph<<<16, 256, 81920, stream>>>(Hs, sadj, qw, qb, kw, kb, maskf);

  k3_gat<<<2048, 256, 74240, stream>>>(
      Htemp, H16, Wgat16, gas, gad, gb, maskf, sng, snb, (float*)d_out);
}
